// Model_40939628265917
// MI455X (gfx1250) — matrix operand order repaired and checked by reading, not compiled, not run on hardware
//
#include <hip/hip_runtime.h>
#include <cstddef>

// ---------- WMMA types (CDNA5 gfx1250, wave32) ----------
typedef __attribute__((ext_vector_type(16))) __bf16 bf16x16;
typedef __attribute__((ext_vector_type(8)))  float  f32x8;

union Frag16 {
    bf16x16      v;
    float4       f[2];
    unsigned int u[8];
};

// round-to-nearest-even fp32 -> bf16 (bits)
__device__ __forceinline__ unsigned short f2bf(float f) {
    unsigned int u = __builtin_bit_cast(unsigned int, f);
    u += 0x7FFFu + ((u >> 16) & 1u);
    return (unsigned short)(u >> 16);
}

#define NEG_INF (-3.0e38f)

// Problem constants
#define BN 2048
#define PP 64
#define DD 512
#define MM 65536
#define HH 1024
#define KK 16
#define SLICES 8           // column slices for fused sim+topk
#define SLICE_COLS (MM / SLICES)   // 8192

// ============================================================
// K1: q = mean_p(patches)  -> bf16  (2048 x 512)
// ============================================================
__global__ __launch_bounds__(256) void kmeanpool(const float* __restrict__ patches,
                                                 unsigned short* __restrict__ qb) {
    int b = blockIdx.x;
    for (int d = threadIdx.x; d < DD; d += 256) {
        const float* p = patches + (size_t)b * PP * DD + d;
        float s = 0.f;
#pragma unroll 8
        for (int k = 0; k < PP; ++k) s += p[(size_t)k * DD];
        qb[(size_t)b * DD + d] = f2bf(s * (1.0f / (float)PP));
    }
}

// ============================================================
// K2: memory fp32 -> bf16 (row-major, 65536 x 512)
// ============================================================
__global__ __launch_bounds__(256) void kconvert(const float* __restrict__ src,
                                                unsigned short* __restrict__ dst,
                                                size_t n) {
    size_t i = (size_t)blockIdx.x * blockDim.x + threadIdx.x;
    size_t stride = (size_t)gridDim.x * blockDim.x;
    for (; i < n; i += stride) dst[i] = f2bf(src[i]);
}

// ============================================================
// K2b: pack W (K x N, row-major fp32) into WMMA-B dwords:
//      Wp[(k>>1)*N + n] = { bf16(W[k+1][n]) , bf16(W[k][n]) }
// ============================================================
__global__ __launch_bounds__(256) void kpack(const float* __restrict__ W,
                                             unsigned int* __restrict__ Wp,
                                             int N) {
    int n  = blockIdx.x * 256 + threadIdx.x;
    int kh = blockIdx.y;                 // k/2
    if (n < N) {
        float lo = W[(size_t)(2 * kh)     * N + n];
        float hi = W[(size_t)(2 * kh + 1) * N + n];
        Wp[(size_t)kh * N + n] = ((unsigned)f2bf(hi) << 16) | (unsigned)f2bf(lo);
    }
}

// ============================================================
// K3: fused sim (q @ memory^T) + per-slice top-16
//     grid (128 row-tiles, 8 slices), 512 threads = 16 waves.
//     Each wave: one 16x16 WMMA tile per iteration (K=512 -> 16 bf16 WMMA).
// ============================================================
__global__ __launch_bounds__(512) void ksim_topk(const unsigned short* __restrict__ qb,
                                                 const unsigned short* __restrict__ memb,
                                                 float* __restrict__ cand_val,
                                                 int*   __restrict__ cand_idx) {
    __shared__ __align__(16) unsigned short qs[16 * DD];   // 16 KB
    __shared__ __align__(16) float simt[16][256];          // 16 KB
    __shared__ float tv[16][16];
    __shared__ int   ti[16][16];

    const int rowtile = blockIdx.x;       // 0..127
    const int slice   = blockIdx.y;       // 0..7
    const int tid  = threadIdx.x;
    const int wave = tid >> 5;            // 0..15
    const int lane = tid & 31;
    const int hi   = lane >> 4;
    const int l16  = lane & 15;

    // stage 16 q rows into LDS (1024 x uint4)
    {
        const uint4* src = (const uint4*)(qb + (size_t)rowtile * 16 * DD);
        uint4* dst = (uint4*)qs;
        for (int i = tid; i < (16 * DD) / 8; i += 512) dst[i] = src[i];
    }
    if (tid < 256) { tv[tid >> 4][tid & 15] = NEG_INF; ti[tid >> 4][tid & 15] = 0; }
    __syncthreads();

    const int c0 = slice * SLICE_COLS;
    const unsigned short* arow = qs + l16 * DD;

    for (int it = 0; it < SLICE_COLS / 256; ++it) {        // 32 iterations
        const int colbase = c0 + it * 256 + wave * 16;
        const unsigned short* bcol = memb + (size_t)(colbase + l16) * DD;

        f32x8 acc = {};
#pragma unroll
        for (int s = 0; s < 16; ++s) {
            Frag16 fa, fb;
            const int ak = s * 32 + hi * 8;        // A: K 0-7 / 8-15 (+16 for upper half)
            fa.f[0] = *(const float4*)(arow + ak);
            fa.f[1] = *(const float4*)(arow + ak + 16);
            const int bk = s * 32 + hi * 8;        // B: K 0-7 / 8-15 (+16 for upper half)
            fb.f[0] = *(const float4*)(bcol + bk);
            fb.f[1] = *(const float4*)(bcol + bk + 16);
            acc = __builtin_amdgcn_wmma_f32_16x16x32_bf16(
                false, fa.v, false, fb.v, (short)0, acc, false, false);
        }

        __syncthreads();                 // prior merge done before we overwrite simt
#pragma unroll
        for (int r = 0; r < 8; ++r)
            simt[r + hi * 8][wave * 16 + l16] = acc[r];
        __syncthreads();

        // wave w maintains top-16 for row w over these 256 columns
        {
            const int row = wave;
            float v0[8];
            const int cbase = lane * 8;
#pragma unroll
            for (int j = 0; j < 8; ++j) v0[j] = simt[row][cbase + j];
            float curmin = tv[row][15];

            while (true) {
                float lm = NEG_INF; int lj = 0;
#pragma unroll
                for (int j = 0; j < 8; ++j) if (v0[j] > lm) { lm = v0[j]; lj = j; }
                float wm = lm; int wl = lane;
                for (int off = 16; off > 0; off >>= 1) {
                    float ov = __shfl_xor(wm, off, 32);
                    int   ol = __shfl_xor(wl, off, 32);
                    if (ov > wm || (ov == wm && ol < wl)) { wm = ov; wl = ol; }
                }
                if (wm <= curmin) break;
                int wj = __shfl(lj, wl, 32);
                if (lane == wl) v0[lj] = NEG_INF;
                int col = c0 + it * 256 + wl * 8 + wj;
                float newmin = curmin;
                if (lane == 0) {
                    int k = 15;
                    while (k > 0 && tv[row][k - 1] < wm) {
                        tv[row][k] = tv[row][k - 1];
                        ti[row][k] = ti[row][k - 1];
                        --k;
                    }
                    tv[row][k] = wm; ti[row][k] = col;
                    newmin = tv[row][15];
                }
                curmin = __shfl(newmin, 0, 32);
            }
        }
    }
    __syncthreads();
    if (tid < 256) {
        int r = tid >> 4, j = tid & 15;
        size_t o = ((size_t)(rowtile * 16 + r)) * (SLICES * 16) + slice * 16 + j;
        cand_val[o] = tv[r][j];
        cand_idx[o] = ti[r][j];
    }
}

// ============================================================
// K4: reduce 8*16 = 128 candidates/row -> final top-16 indices
// ============================================================
__global__ __launch_bounds__(32) void kfinal_topk(const float* __restrict__ cand_val,
                                                  const int*   __restrict__ cand_idx,
                                                  int* __restrict__ fidx) {
    const int row = blockIdx.x, lane = threadIdx.x;
    float v[4]; int c[4];
#pragma unroll
    for (int j = 0; j < 4; ++j) {
        size_t o = (size_t)row * 128 + lane * 4 + j;
        v[j] = cand_val[o]; c[j] = cand_idx[o];
    }
    for (int k = 0; k < KK; ++k) {
        float lm = v[0]; int lj = 0;
#pragma unroll
        for (int j = 1; j < 4; ++j) if (v[j] > lm) { lm = v[j]; lj = j; }
        float wm = lm; int wl = lane;
        for (int off = 16; off > 0; off >>= 1) {
            float ov = __shfl_xor(wm, off, 32);
            int   ol = __shfl_xor(wl, off, 32);
            if (ov > wm || (ov == wm && ol < wl)) { wm = ov; wl = ol; }
        }
        int col = __shfl(c[lj], wl, 32);
        if (lane == wl) v[lj] = NEG_INF;
        if (lane == 0) fidx[(size_t)row * KK + k] = col;
    }
}

// ============================================================
// K6: fused MLP + mean_k + residual.  One block (8 waves) per sample.
//     GEMM1: (16x512)@(512x1024) + b1, exact GELU -> bf16 LDS
//     GEMM2: (16x1024)@(1024x512), mean over 16 rows, + b2
//     out = patches + broadcast(lmean)
// ============================================================
__global__ __launch_bounds__(256) void kmlp(const unsigned short* __restrict__ memb,
                                            const int*   __restrict__ fidx,
                                            const unsigned int* __restrict__ w1p,
                                            const float* __restrict__ b1,
                                            const unsigned int* __restrict__ w2p,
                                            const float* __restrict__ b2,
                                            const float* __restrict__ patches,
                                            float* __restrict__ out) {
    __shared__ __align__(16) unsigned short ret[16 * DD];    // 16 KB
    __shared__ __align__(16) unsigned short hbuf[16 * HH];   // 32 KB
    __shared__ float lmean[DD];                              // 2 KB

    const int b = blockIdx.x;
    const int tid = threadIdx.x;
    const int wave = tid >> 5, lane = tid & 31;
    const int hi = lane >> 4, l16 = lane & 15;

    // gather 16 retrieved memory rows (bf16) into LDS
    for (int r = wave; r < 16; r += 8) {
        int idx = fidx[(size_t)b * KK + r];
        const uint4* src = (const uint4*)(memb + (size_t)idx * DD);
        uint4* dst = (uint4*)(ret + r * DD);
        for (int i = lane; i < DD / 8; i += 32) dst[i] = src[i];
    }
    __syncthreads();

    // GEMM1 + bias + exact GELU
    {
        const unsigned short* arow = ret + l16 * DD;
        for (int nt = wave; nt < HH / 16; nt += 8) {
            const int n = nt * 16 + l16;
            f32x8 acc = {};
#pragma unroll
            for (int s = 0; s < DD / 32; ++s) {
                Frag16 fa, fb;
                const int ak = s * 32 + hi * 16;
                fa.f[0] = *(const float4*)(arow + ak);
                fa.f[1] = *(const float4*)(arow + ak + 8);
                const int kb2 = s * 16 + hi * 8;   // dword row in packed W1
#pragma unroll
                for (int j = 0; j < 8; ++j) fb.u[j] = w1p[(size_t)(kb2 + j) * HH + n];
                acc = __builtin_amdgcn_wmma_f32_16x16x32_bf16(
                    false, fa.v, false, fb.v, (short)0, acc, false, false);
            }
            const float bias = b1[n];
#pragma unroll
            for (int r = 0; r < 8; ++r) {
                float x = acc[r] + bias;
                float g = 0.5f * x * (1.0f + erff(x * 0.70710678118654752440f));
                hbuf[(size_t)(r + hi * 8) * HH + n] = f2bf(g);
            }
        }
    }
    __syncthreads();

    // GEMM2 + mean over the 16 retrieved rows + b2
    {
        const unsigned short* arow = hbuf + l16 * HH;
        for (int nt = wave; nt < DD / 16; nt += 8) {
            const int n = nt * 16 + l16;
            f32x8 acc = {};
#pragma unroll
            for (int s = 0; s < HH / 32; ++s) {
                Frag16 fa, fb;
                const int ak = s * 32 + hi * 16;
                fa.f[0] = *(const float4*)(arow + ak);
                fa.f[1] = *(const float4*)(arow + ak + 8);
                const int kb2 = s * 16 + hi * 8;   // dword row in packed W2
#pragma unroll
                for (int j = 0; j < 8; ++j) fb.u[j] = w2p[(size_t)(kb2 + j) * DD + n];
                acc = __builtin_amdgcn_wmma_f32_16x16x32_bf16(
                    false, fa.v, false, fb.v, (short)0, acc, false, false);
            }
            float s8 = 0.f;
#pragma unroll
            for (int r = 0; r < 8; ++r) s8 += acc[r];      // rows (hi*8 .. hi*8+7)
            s8 += __shfl_xor(s8, 16, 32);                  // combine the two row halves
            if (hi == 0) lmean[n] = s8 * (1.0f / (float)KK) + b2[n];
        }
    }
    __syncthreads();

    // residual broadcast: out[b][p][d] = patches[b][p][d] + lmean[d]
    const float4* pin  = (const float4*)(patches + (size_t)b * PP * DD);
    float4*       pout = (float4*)(out + (size_t)b * PP * DD);
    for (int i = tid; i < (PP * DD) / 4; i += 256) {
        const int d4 = (i & (DD / 4 - 1)) * 4;
        float4 v = pin[i];
        v.x += lmean[d4 + 0];
        v.y += lmean[d4 + 1];
        v.z += lmean[d4 + 2];
        v.w += lmean[d4 + 3];
        pout[i] = v;
    }
}

// ============================================================
extern "C" void kernel_launch(void* const* d_in, const int* in_sizes, int n_in,
                              void* d_out, int out_size, void* d_ws, size_t ws_size,
                              hipStream_t stream) {
    const float* patches = (const float*)d_in[0];
    const float* memory  = (const float*)d_in[1];
    const float* W1      = (const float*)d_in[2];
    const float* b1      = (const float*)d_in[3];
    const float* W2      = (const float*)d_in[4];
    const float* b2      = (const float*)d_in[5];
    // d_in[6] = top_k (fixed at 16)

    char* ws = (char*)d_ws;
    unsigned short* qb   = (unsigned short*)ws;  ws += (size_t)BN * DD * 2;           // 2 MB
    unsigned short* memb = (unsigned short*)ws;  ws += (size_t)MM * DD * 2;           // 64 MB
    unsigned int*   w1p  = (unsigned int*)ws;    ws += (size_t)(DD / 2) * HH * 4;     // 1 MB
    unsigned int*   w2p  = (unsigned int*)ws;    ws += (size_t)(HH / 2) * DD * 4;     // 1 MB
    float* cand_val      = (float*)ws;           ws += (size_t)BN * 128 * 4;          // 1 MB
    int*   cand_idx      = (int*)ws;             ws += (size_t)BN * 128 * 4;          // 1 MB
    int*   fidx          = (int*)ws;             ws += (size_t)BN * KK * 4;

    kmeanpool<<<BN, 256, 0, stream>>>(patches, qb);
    kconvert <<<4096, 256, 0, stream>>>(memory, memb, (size_t)MM * DD);
    kpack    <<<dim3(HH / 256, DD / 2), 256, 0, stream>>>(W1, w1p, HH);
    kpack    <<<dim3(DD / 256, HH / 2), 256, 0, stream>>>(W2, w2p, DD);
    ksim_topk<<<dim3(BN / 16, SLICES), 512, 0, stream>>>(qb, memb, cand_val, cand_idx);
    kfinal_topk<<<BN, 32, 0, stream>>>(cand_val, cand_idx, fidx);
    kmlp     <<<BN, 256, 0, stream>>>(memb, fidx, w1p, b1, w2p, b2, patches, (float*)d_out);
}
